// SSMPathway_6828998000924
// MI455X (gfx1250) — matrix operand order repaired and checked by reading, compiled, not run on hardware
//
#include <hip/hip_runtime.h>
#include <hip/hip_bf16.h>
#include <math.h>

#define D_MODEL 1024
#define D_INNER 2048
#define D_STATE 16
#define DT_RANK 64
#define CONV_K 4
#define B_SZ 2
#define SEQ 2048
#define TOKENS (B_SZ * SEQ)   // 4096
#define XPROJ_N (DT_RANK + 2 * D_STATE)  // 96

typedef __attribute__((ext_vector_type(16))) __bf16 v16bf;
typedef __attribute__((ext_vector_type(8)))  __bf16 v8bf;
typedef __attribute__((ext_vector_type(8)))  float  v8f;

__device__ __forceinline__ __bf16 f2bf(float f) {
    unsigned u = __builtin_bit_cast(unsigned, f);
    u += 0x7FFFu + ((u >> 16) & 1u);          // round-to-nearest-even
    unsigned short h = (unsigned short)(u >> 16);
    return __builtin_bit_cast(__bf16, h);
}

__device__ __forceinline__ float sigmoidf_(float v) { return 1.0f / (1.0f + __expf(-v)); }

// Build a 16-element bf16 fragment from two 16-byte (b128) loads `gap` elements apart.
__device__ __forceinline__ v16bf load_frag(const __bf16* __restrict__ p, int gap) {
    v8bf lo = *(const v8bf*)(p);
    v8bf hi = *(const v8bf*)(p + gap);
    v16bf f;
#pragma unroll
    for (int i = 0; i < 8; ++i) { f[i] = lo[i]; f[i + 8] = hi[i]; }
    return f;
}

// ---------------------------------------------------------------------------
// LayerNorm over d_model, emit bf16 A-operand for the in-proj GEMM.
// ---------------------------------------------------------------------------
__global__ void ln_bf16_kernel(const float* __restrict__ x,
                               const float* __restrict__ g,
                               const float* __restrict__ b,
                               __bf16* __restrict__ xn) {
    const int t = blockIdx.x;
    const int tid = threadIdx.x;
    const float* xr = x + (size_t)t * D_MODEL;
    float s = 0.f, s2 = 0.f;
    for (int i = tid; i < D_MODEL; i += 256) { float v = xr[i]; s += v; s2 += v * v; }
    __shared__ float r1[256], r2[256];
    r1[tid] = s; r2[tid] = s2; __syncthreads();
    for (int w = 128; w > 0; w >>= 1) {
        if (tid < w) { r1[tid] += r1[tid + w]; r2[tid] += r2[tid + w]; }
        __syncthreads();
    }
    const float mu = r1[0] * (1.0f / D_MODEL);
    const float var = r2[0] * (1.0f / D_MODEL) - mu * mu;
    const float rstd = rsqrtf(var + 1e-5f);
    for (int i = tid; i < D_MODEL; i += 256) {
        float v = (xr[i] - mu) * rstd * g[i] + b[i];
        xn[(size_t)t * D_MODEL + i] = f2bf(v);
    }
}

// ---------------------------------------------------------------------------
// f32 -> bf16 converters
// ---------------------------------------------------------------------------
__global__ void cvt_bf16_kernel(const float* __restrict__ in, __bf16* __restrict__ out, int n) {
    int i = blockIdx.x * blockDim.x + threadIdx.x;
    if (i < n) out[i] = f2bf(in[i]);
}

__global__ void cvt_bf16_strided_kernel(const float* __restrict__ in, int ld,
                                        __bf16* __restrict__ out, int cols, int n) {
    int i = blockIdx.x * blockDim.x + threadIdx.x;
    if (i < n) {
        int r = i / cols, c = i - r * cols;
        out[i] = f2bf(in[(size_t)r * ld + c]);
    }
}

// ---------------------------------------------------------------------------
// WMMA GEMM: C[M,N] = A[M,K](bf16) * W[N,K]^T(bf16)  (+bias, act, +residual)
// One wave computes a 32x64 strip: MT=2 M-tiles x NT=4 N-tiles (8 accumulators).
// K is unrolled by 64 with ping-pong fragment buffers (no register copies):
// buffer X is consumed by WMMA while buffer Y's loads are in flight.
// Requires K % 64 == 0 (true for all call sites: 1024 / 2048 / 64).
// amdgpu_waves_per_eu(1) lifts the occupancy-driven VGPR cap so the ~220 live
// VGPRs (of the 1024/wave architectural budget) allocate without spills.
// act: 0 = none, 1 = softplus
// ---------------------------------------------------------------------------
#define GEMM_MT 2
#define GEMM_NT 4
__global__ void
__attribute__((amdgpu_flat_work_group_size(256, 256), amdgpu_waves_per_eu(1)))
gemm_wmma_kernel(const __bf16* __restrict__ A, int lda,
                 const __bf16* __restrict__ W, int ldb,
                 float* __restrict__ C, int ldc,
                 int M, int N, int K,
                 const float* __restrict__ bias,
                 const float* __restrict__ residual,
                 int act) {
    const int lane = threadIdx.x & 31;
    const int gw = blockIdx.x * (blockDim.x >> 5) + (threadIdx.x >> 5);
    const int tilesM = M >> 5;                 // 32 rows per wave
    const int tilesN = (N + 63) >> 6;          // 64 cols per wave
    if (gw >= tilesM * tilesN) return;
    const int tm = gw % tilesM;
    const int tn = gw / tilesM;
    const int m0 = tm << 5;
    const int n0 = tn << 6;

    const int l15   = lane & 15;
    const int hi    = lane >> 4;        // 0: lanes 0-15, 1: lanes 16-31
    const int ka    = hi << 4;          // A fragment K sub-offset 0 / 16
    const int kbOff = hi << 4;          // B fragment K sub-offset 0 / 16

    const __bf16* Abase[GEMM_MT];
#pragma unroll
    for (int mt = 0; mt < GEMM_MT; ++mt)
        Abase[mt] = A + (size_t)(m0 + (mt << 4) + l15) * lda + ka;
    const __bf16* Wbase[GEMM_NT];
#pragma unroll
    for (int t = 0; t < GEMM_NT; ++t) {
        int ncol = n0 + (t << 4) + l15;
        int nc = (ncol < N) ? ncol : 0;        // clamp loads for partial N tiles
        Wbase[t] = W + (size_t)nc * ldb + kbOff;
    }

    v8f acc[GEMM_MT][GEMM_NT];
#pragma unroll
    for (int mt = 0; mt < GEMM_MT; ++mt)
#pragma unroll
        for (int t = 0; t < GEMM_NT; ++t)
#pragma unroll
            for (int i = 0; i < 8; ++i) acc[mt][t][i] = 0.f;

    // ---- ping-pong pipeline over 64 K-elements per iteration ----
    v16bf aP[GEMM_MT], bP[GEMM_NT];   // "ping" buffer
    v16bf aQ[GEMM_MT], bQ[GEMM_NT];   // "pong" buffer
#pragma unroll
    for (int mt = 0; mt < GEMM_MT; ++mt) aP[mt] = load_frag(Abase[mt], 8);
#pragma unroll
    for (int t = 0; t < GEMM_NT; ++t)    bP[t]  = load_frag(Wbase[t], 8);

    for (int k0 = 0; k0 < K; k0 += 64) {
        const int k1 = k0 + 32;
        // issue pong loads, then consume ping with 8 WMMAs
#pragma unroll
        for (int mt = 0; mt < GEMM_MT; ++mt) aQ[mt] = load_frag(Abase[mt] + k1, 8);
#pragma unroll
        for (int t = 0; t < GEMM_NT; ++t)    bQ[t]  = load_frag(Wbase[t] + k1, 8);
#pragma unroll
        for (int mt = 0; mt < GEMM_MT; ++mt)
#pragma unroll
            for (int t = 0; t < GEMM_NT; ++t)
                acc[mt][t] = __builtin_amdgcn_wmma_f32_16x16x32_bf16(
                    false, aP[mt], false, bP[t], (short)0, acc[mt][t], false, false);

        const int k2 = k0 + 64;
        if (k2 < K) {
            // issue next ping loads, then consume pong
#pragma unroll
            for (int mt = 0; mt < GEMM_MT; ++mt) aP[mt] = load_frag(Abase[mt] + k2, 8);
#pragma unroll
            for (int t = 0; t < GEMM_NT; ++t)    bP[t]  = load_frag(Wbase[t] + k2, 8);
        }
#pragma unroll
        for (int mt = 0; mt < GEMM_MT; ++mt)
#pragma unroll
            for (int t = 0; t < GEMM_NT; ++t)
                acc[mt][t] = __builtin_amdgcn_wmma_f32_16x16x32_bf16(
                    false, aQ[mt], false, bQ[t], (short)0, acc[mt][t], false, false);
    }

    // Epilogue: C/D layout — VGPR r: row = tile_m0 + r + (hi?8:0), col = tile_n0 + l15
#pragma unroll
    for (int mt = 0; mt < GEMM_MT; ++mt) {
        const int rbase = m0 + (mt << 4) + (hi << 3);
#pragma unroll
        for (int t = 0; t < GEMM_NT; ++t) {
            int ncol = n0 + (t << 4) + l15;
            if (ncol >= N) continue;
            float bv = bias ? bias[ncol] : 0.f;
#pragma unroll
            for (int r = 0; r < 8; ++r) {
                float v = acc[mt][t][r] + bv;
                if (act == 1) v = (v > 20.f) ? v : log1pf(__expf(v));
                size_t off = (size_t)(rbase + r) * ldc + ncol;
                if (residual) v += residual[off];
                C[off] = v;
            }
        }
    }
}

// ---------------------------------------------------------------------------
// Causal depthwise conv1d (k=4) + bias + SiLU over the xc half of xz.
// ---------------------------------------------------------------------------
__global__ void conv_silu_kernel(const float* __restrict__ xz,
                                 const float* __restrict__ cw,
                                 const float* __restrict__ cb,
                                 float* __restrict__ xc,
                                 __bf16* __restrict__ xc_bf) {
    int idx = blockIdx.x * blockDim.x + threadIdx.x;
    if (idx >= B_SZ * SEQ * D_INNER) return;
    int d = idx & (D_INNER - 1);
    int l = (idx >> 11) & (SEQ - 1);
    int b = idx >> 22;
    float acc = cb[d];
#pragma unroll
    for (int j = 0; j < CONV_K; ++j) {
        int ls = l - (CONV_K - 1) + j;
        if (ls >= 0)
            acc = fmaf(cw[d * CONV_K + j],
                       xz[((size_t)(b * SEQ + ls)) * (2 * D_INNER) + d], acc);
    }
    float v = acc * sigmoidf_(acc);
    size_t o = (size_t)(b * SEQ + l) * D_INNER + d;
    xc[o] = v;
    xc_bf[o] = f2bf(v);
}

// ---------------------------------------------------------------------------
// Selective scan, sequential in L. One thread owns one d-channel: 16-state in
// registers. B/C (shared across d) staged in LDS per step. Fuses +u*D and
// *silu(z) gating; emits bf16 A-operand for the out-proj GEMM.
// ---------------------------------------------------------------------------
__global__ void scan_kernel(const float* __restrict__ delta,
                            const float* __restrict__ xc,
                            const float* __restrict__ xdbl,
                            const float* __restrict__ xz,
                            const float* __restrict__ A_log,
                            const float* __restrict__ Dp,
                            __bf16* __restrict__ y_bf) {
    const int b = blockIdx.x >> 3;
    const int d = ((blockIdx.x & 7) << 8) + threadIdx.x;
    float Arow[D_STATE], s[D_STATE];
#pragma unroll
    for (int n = 0; n < D_STATE; ++n) {
        Arow[n] = -__expf(A_log[(size_t)d * D_STATE + n]);
        s[n] = 0.f;
    }
    const float Dd = Dp[d];
    __shared__ float sB[D_STATE], sC[D_STATE];

    for (int l = 0; l < SEQ; ++l) {
        const size_t t = (size_t)b * SEQ + l;
        __syncthreads();   // protect previous step's LDS reads
        if (threadIdx.x < D_STATE)
            sB[threadIdx.x] = xdbl[t * XPROJ_N + DT_RANK + threadIdx.x];
        else if (threadIdx.x < 2 * D_STATE)
            sC[threadIdx.x - D_STATE] = xdbl[t * XPROJ_N + DT_RANK + D_STATE + (threadIdx.x - D_STATE)];
        __syncthreads();

        const float dl = delta[t * D_INNER + d];
        const float u  = xc[t * D_INNER + d];
        const float du = dl * u;
        float y = 0.f;
#pragma unroll
        for (int n = 0; n < D_STATE; ++n) {
            float dA = __expf(dl * Arow[n]);
            s[n] = fmaf(dA, s[n], du * sB[n]);
            y = fmaf(s[n], sC[n], y);
        }
        const float zv = xz[t * (2 * D_INNER) + D_INNER + d];
        const float out = (y + u * Dd) * (zv * sigmoidf_(zv));
        y_bf[t * D_INNER + d] = f2bf(out);
    }
}

// ---------------------------------------------------------------------------
extern "C" void kernel_launch(void* const* d_in, const int* in_sizes, int n_in,
                              void* d_out, int out_size, void* d_ws, size_t ws_size,
                              hipStream_t stream) {
    const float* x       = (const float*)d_in[0];
    const float* ln_g    = (const float*)d_in[1];
    const float* ln_b    = (const float*)d_in[2];
    const float* W_in    = (const float*)d_in[3];
    const float* conv_w  = (const float*)d_in[4];
    const float* conv_b  = (const float*)d_in[5];
    const float* A_log   = (const float*)d_in[6];
    const float* D_param = (const float*)d_in[7];
    const float* x_projw = (const float*)d_in[8];
    const float* dt_w    = (const float*)d_in[9];
    const float* dt_b    = (const float*)d_in[10];
    const float* W_out   = (const float*)d_in[11];
    float* out = (float*)d_out;

    char* ws = (char*)d_ws;
    size_t off = 0;
    auto alloc = [&](size_t bytes) -> void* {
        void* p = ws + off;
        off = (off + bytes + 255) & ~(size_t)255;
        return p;
    };

    __bf16* xn_bf    = (__bf16*)alloc((size_t)TOKENS * D_MODEL * 2);
    __bf16* Win_bf   = (__bf16*)alloc((size_t)2 * D_INNER * D_MODEL * 2);
    __bf16* Wout_bf  = (__bf16*)alloc((size_t)D_MODEL * D_INNER * 2);
    __bf16* Xp_bf    = (__bf16*)alloc((size_t)XPROJ_N * D_INNER * 2);
    __bf16* Dtw_bf   = (__bf16*)alloc((size_t)D_INNER * DT_RANK * 2);
    float*  xz       = (float*) alloc((size_t)TOKENS * 2 * D_INNER * 4);
    float*  xc       = (float*) alloc((size_t)TOKENS * D_INNER * 4);
    __bf16* xc_bf    = (__bf16*)alloc((size_t)TOKENS * D_INNER * 2);
    float*  x_dbl    = (float*) alloc((size_t)TOKENS * XPROJ_N * 4);
    __bf16* dr_bf    = (__bf16*)alloc((size_t)TOKENS * DT_RANK * 2);
    float*  delta    = (float*) alloc((size_t)TOKENS * D_INNER * 4);
    __bf16* y_bf     = (__bf16*)alloc((size_t)TOKENS * D_INNER * 2);

    // 1) LayerNorm -> bf16
    ln_bf16_kernel<<<TOKENS, 256, 0, stream>>>(x, ln_g, ln_b, xn_bf);

    // 2) Weight conversions -> bf16
    {
        int n1 = 2 * D_INNER * D_MODEL;
        cvt_bf16_kernel<<<(n1 + 255) / 256, 256, 0, stream>>>(W_in, Win_bf, n1);
        int n2 = D_MODEL * D_INNER;
        cvt_bf16_kernel<<<(n2 + 255) / 256, 256, 0, stream>>>(W_out, Wout_bf, n2);
        int n3 = XPROJ_N * D_INNER;
        cvt_bf16_kernel<<<(n3 + 255) / 256, 256, 0, stream>>>(x_projw, Xp_bf, n3);
        int n4 = D_INNER * DT_RANK;
        cvt_bf16_kernel<<<(n4 + 255) / 256, 256, 0, stream>>>(dt_w, Dtw_bf, n4);
    }

    // 3) in-proj GEMM: xz[T,4096] = xn * W_in^T
    {
        int waves = (TOKENS / 32) * ((2 * D_INNER + 63) / 64);
        gemm_wmma_kernel<<<(waves + 7) / 8, 256, 0, stream>>>(
            xn_bf, D_MODEL, Win_bf, D_MODEL, xz, 2 * D_INNER,
            TOKENS, 2 * D_INNER, D_MODEL, nullptr, nullptr, 0);
    }

    // 4) depthwise conv + SiLU
    {
        int n = B_SZ * SEQ * D_INNER;
        conv_silu_kernel<<<(n + 255) / 256, 256, 0, stream>>>(xz, conv_w, conv_b, xc, xc_bf);
    }

    // 5) x_proj GEMM: x_dbl[T,96] = xc * x_proj_w^T
    {
        int waves = (TOKENS / 32) * ((XPROJ_N + 63) / 64);
        gemm_wmma_kernel<<<(waves + 7) / 8, 256, 0, stream>>>(
            xc_bf, D_INNER, Xp_bf, D_INNER, x_dbl, XPROJ_N,
            TOKENS, XPROJ_N, D_INNER, nullptr, nullptr, 0);
    }

    // 6) delta_r slice -> bf16, then dt GEMM with fused bias + softplus
    {
        int n = TOKENS * DT_RANK;
        cvt_bf16_strided_kernel<<<(n + 255) / 256, 256, 0, stream>>>(
            x_dbl, XPROJ_N, dr_bf, DT_RANK, n);
        int waves = (TOKENS / 32) * ((D_INNER + 63) / 64);
        gemm_wmma_kernel<<<(waves + 7) / 8, 256, 0, stream>>>(
            dr_bf, DT_RANK, Dtw_bf, DT_RANK, delta, D_INNER,
            TOKENS, D_INNER, DT_RANK, dt_b, nullptr, 1);
    }

    // 7) selective scan (fused +u*D and silu(z) gating) -> bf16
    scan_kernel<<<B_SZ * (D_INNER / 256), 256, 0, stream>>>(
        delta, xc, x_dbl, xz, A_log, D_param, y_bf);

    // 8) out-proj GEMM with fused residual add -> d_out
    {
        int waves = (TOKENS / 32) * ((D_MODEL + 63) / 64);
        gemm_wmma_kernel<<<(waves + 7) / 8, 256, 0, stream>>>(
            y_bf, D_INNER, Wout_bf, D_INNER, out, D_MODEL,
            TOKENS, D_MODEL, D_INNER, nullptr, x, 0);
    }
    (void)in_sizes; (void)n_in; (void)out_size; (void)ws_size;
}
